// ProtBertStrEC_31868657336853
// MI455X (gfx1250) — matrix operand order repaired and checked by reading, not compiled, not run on hardware
//
#include <hip/hip_runtime.h>
#include <hip/hip_bf16.h>
#include <stdint.h>

// ---------------------------------------------------------------------------
// CDNA5 (gfx1250) types for WMMA / packed bf16 VALU
// ---------------------------------------------------------------------------
typedef __attribute__((ext_vector_type(16))) __bf16 v16bf;
typedef __attribute__((ext_vector_type(8)))  float  v8f;
typedef __attribute__((ext_vector_type(2)))  __bf16 bf16x2;

union Frag16 { v16bf v; uint4 q[2]; };

__device__ __forceinline__ float bf2f(unsigned short u) {
    union { unsigned int i; float f; } x; x.i = ((unsigned int)u) << 16; return x.f;
}
__device__ __forceinline__ unsigned short f2bf(float f) {
    union { float f; unsigned int i; } x; x.f = f;
    unsigned int u = x.i;
    u += 0x7FFFu + ((u >> 16) & 1u);            // round-to-nearest-even
    return (unsigned short)(u >> 16);
}

// packed bf16: |a-b| and a*b on 2 lanes at once (v_pk_add_bf16 / v_pk_mul_bf16)
__device__ __forceinline__ unsigned int pk_abs_diff(unsigned int a, unsigned int b) {
    bf16x2 x = __builtin_bit_cast(bf16x2, a);
    bf16x2 y = __builtin_bit_cast(bf16x2, b);
    bf16x2 d = x - y;
    return __builtin_bit_cast(unsigned int, d) & 0x7FFF7FFFu;  // packed abs
}
__device__ __forceinline__ unsigned int pk_mul(unsigned int a, unsigned int b) {
    bf16x2 x = __builtin_bit_cast(bf16x2, a);
    bf16x2 y = __builtin_bit_cast(bf16x2, b);
    bf16x2 m = x * y;
    return __builtin_bit_cast(unsigned int, m);
}

#define L_SEQ   256
#define C_DIM   1024
#define NEC_DIM 2000

// ---------------------------------------------------------------------------
// 1) mask: s = sum(attention_mask); zero position 0 and position s
// ---------------------------------------------------------------------------
__global__ void k_mask(const int* __restrict__ am, int* __restrict__ mask_eff) {
    __shared__ int red[256];
    int t = threadIdx.x;
    red[t] = am[t];
    __syncthreads();
    for (int off = 128; off > 0; off >>= 1) {
        if (t < off) red[t] += red[t + off];
        __syncthreads();
    }
    int s = red[0];
    mask_eff[t] = (t == 0 || t == s) ? 0 : am[t];
}

// ---------------------------------------------------------------------------
// 2) zh[l][c] = bf16(attentions[l][c] * mask[l])   ([L][C], row-major, k-contig)
// ---------------------------------------------------------------------------
__global__ void k_prep_z(const float* __restrict__ att, const int* __restrict__ mask_eff,
                         unsigned short* __restrict__ zh) {
    int l = blockIdx.x, t = threadIdx.x;
    float msk = (float)mask_eff[l];
    float4 v = ((const float4*)(att + (size_t)l * C_DIM))[t];
    uint2 pk;
    pk.x = (unsigned int)f2bf(v.x * msk) | ((unsigned int)f2bf(v.y * msk) << 16);
    pk.y = (unsigned int)f2bf(v.z * msk) | ((unsigned int)f2bf(v.w * msk) << 16);
    ((uint2*)(zh + (size_t)l * C_DIM))[t] = pk;
}

// ---------------------------------------------------------------------------
// 3) conv1_w f32 -> bf16 (layout unchanged: [1024 o][2048 c])
// ---------------------------------------------------------------------------
__global__ void k_prep_w(const float* __restrict__ w, unsigned short* __restrict__ wb, int n4) {
    int idx = blockIdx.x * blockDim.x + threadIdx.x;
    if (idx >= n4) return;
    float4 v = ((const float4*)w)[idx];
    uint2 pk;
    pk.x = (unsigned int)f2bf(v.x) | ((unsigned int)f2bf(v.y) << 16);
    pk.y = (unsigned int)f2bf(v.z) | ((unsigned int)f2bf(v.w) << 16);
    ((uint2*)wb)[idx] = pk;
}

// ---------------------------------------------------------------------------
// 4) GEMV: one wave (32 lanes) per output row, coalesced, shuffle-reduce
// ---------------------------------------------------------------------------
__global__ void k_gemv(const float* __restrict__ W, const float* __restrict__ x,
                       const float* __restrict__ b, float* __restrict__ y,
                       int nOut, int K) {
    int wave = threadIdx.x >> 5, lane = threadIdx.x & 31;
    int o = blockIdx.x * (blockDim.x >> 5) + wave;
    if (o >= nOut) return;
    const float* row = W + (size_t)o * K;
    float s = 0.f;
    for (int c = lane; c < K; c += 32) s += row[c] * x[c];
    for (int off = 16; off > 0; off >>= 1) s += __shfl_xor(s, off, 32);
    if (lane == 0) y[o] = s + b[o];
}

// ---------------------------------------------------------------------------
// 5) Pairwise-feature GEMM via WMMA bf16:
//    h[o, i, j] = relu( sum_c w[o,c]|z[c,j]-z[c,i]| + sum_c w[o,1024+c] z[c,i]z[c,j] + b[o] )
//    M=1024 N=65536 K=2048, bf16 inputs, f32 accumulate, bf16 h output.
//    WG tile: 256 o x 64 px (8x8 patch); wave: 2 M-tiles x 4 N-tiles.
// ---------------------------------------------------------------------------
__global__ __launch_bounds__(256)
void k_pairgemm(const unsigned short* __restrict__ zh,
                const unsigned short* __restrict__ cw,
                const float* __restrict__ bias,
                unsigned short* __restrict__ hbuf) {
    // padded rows (40 ushorts = 80B, 16B-aligned, conflict-free b128 DS access)
    __shared__ unsigned short fd[64][40];   // |z_j - z_i| tile: [pixel][k]
    __shared__ unsigned short fm[64][40];   // z_i * z_j  tile: [pixel][k]

    const int tid  = threadIdx.x;
    const int lane = tid & 31;
    const int wave = tid >> 5;
    const int j0 = blockIdx.x * 8;
    const int i0 = blockIdx.y * 8;
    const int o0 = blockIdx.z * 256 + wave * 16;   // second M-tile at o0+128

    v8f acc[2][4] = {};

    // feat-builder assignment: 4 threads per pixel, 8 k each
    const int pB  = tid >> 2;            // pixel 0..63 in 8x8 patch
    const int kkB = (tid & 3) * 8;       // k sub-offset 0/8/16/24
    const unsigned short* zrowi = zh + (size_t)(i0 + (pB >> 3)) * C_DIM + kkB;
    const unsigned short* zrowj = zh + (size_t)(j0 + (pB & 7)) * C_DIM + kkB;

    // A-fragment addressing (16-bit A 16x32 layout: lane m = o-row,
    // elements 0..7 -> K=kb..kb+7, elements 8..15 -> K=kb+16..kb+23, kb=8*(lane>>4))
    const int m  = lane & 15;
    const int kb = (lane >> 4) * 8;
    const unsigned short* arow0 = cw + (size_t)(o0 + m) * 2048 + kb;
    const unsigned short* arow1 = cw + (size_t)(o0 + 128 + m) * 2048 + kb;

    // B-fragment addressing (16-bit B 32x16: lane col = n,
    // elements 0..7 -> K=kB..kB+7, elements 8..15 -> K=kB+16..kB+23, kB=8*(lane>>4))
    const int kB = (lane >> 4) * 8;

    for (int k0 = 0; k0 < 1024; k0 += 32) {
        // ---- cooperatively build bf16 feature tiles in LDS (packed bf16 VALU) ----
        union { uint4 q; unsigned int u[4]; } qi, qj, qd, qm;
        qi.q = *(const uint4*)(zrowi + k0);
        qj.q = *(const uint4*)(zrowj + k0);
#pragma unroll
        for (int e = 0; e < 4; ++e) {
            qd.u[e] = pk_abs_diff(qi.u[e], qj.u[e]);
            qm.u[e] = pk_mul(qi.u[e], qj.u[e]);
        }
        *(uint4*)&fd[pB][kkB] = qd.q;
        *(uint4*)&fm[pB][kkB] = qm.q;
        __syncthreads();

        // ---- A fragments: conv1_w bf16, diff + mul halves, two M-tiles ----
        __builtin_prefetch(arow0 + k0 + 256, 0, 1);   // global_prefetch_b8
        __builtin_prefetch(arow1 + k0 + 256, 0, 1);
        Frag16 A0d, A0m, A1d, A1m;
        A0d.q[0] = *(const uint4*)(arow0 + k0);
        A0d.q[1] = *(const uint4*)(arow0 + k0 + 16);
        A0m.q[0] = *(const uint4*)(arow0 + 1024 + k0);
        A0m.q[1] = *(const uint4*)(arow0 + 1024 + k0 + 16);
        A1d.q[0] = *(const uint4*)(arow1 + k0);
        A1d.q[1] = *(const uint4*)(arow1 + k0 + 16);
        A1m.q[0] = *(const uint4*)(arow1 + 1024 + k0);
        A1m.q[1] = *(const uint4*)(arow1 + 1024 + k0 + 16);

        // ---- 4 pixel sub-tiles: B from LDS feeds 4 WMMAs each ----
#pragma unroll
        for (int nt = 0; nt < 4; ++nt) {
            const int p = nt * 16 + m;
            Frag16 Bd, Bm;
            Bd.q[0] = *(const uint4*)&fd[p][kB];
            Bd.q[1] = *(const uint4*)&fd[p][kB + 16];
            Bm.q[0] = *(const uint4*)&fm[p][kB];
            Bm.q[1] = *(const uint4*)&fm[p][kB + 16];
            acc[0][nt] = __builtin_amdgcn_wmma_f32_16x16x32_bf16(
                false, A0d.v, false, Bd.v, (short)0, acc[0][nt], false, false);
            acc[0][nt] = __builtin_amdgcn_wmma_f32_16x16x32_bf16(
                false, A0m.v, false, Bm.v, (short)0, acc[0][nt], false, false);
            acc[1][nt] = __builtin_amdgcn_wmma_f32_16x16x32_bf16(
                false, A1d.v, false, Bd.v, (short)0, acc[1][nt], false, false);
            acc[1][nt] = __builtin_amdgcn_wmma_f32_16x16x32_bf16(
                false, A1m.v, false, Bm.v, (short)0, acc[1][nt], false, false);
        }
        __syncthreads();
    }

    // ---- epilogue: bias + ReLU, store bf16 h[o][i][j] ----
    // f32 C/D layout: VGPR r -> M = r + 8*(lane>>4); N = lane&15
    const int hi  = lane >> 4;
    const int col = lane & 15;
#pragma unroll
    for (int mt = 0; mt < 2; ++mt) {
        const int ob = o0 + mt * 128;
#pragma unroll
        for (int nt = 0; nt < 4; ++nt) {
            const int p  = nt * 16 + col;
            const size_t pix = (size_t)(i0 + (p >> 3)) * 256 + (j0 + (p & 7));
#pragma unroll
            for (int r = 0; r < 8; ++r) {
                const int o = ob + hi * 8 + r;
                float v = acc[mt][nt][r] + bias[o];
                hbuf[(size_t)o * 65536 + pix] = f2bf(fmaxf(v, 0.f));
            }
        }
    }
}

// ---------------------------------------------------------------------------
// 6) zero accumulation buffer
// ---------------------------------------------------------------------------
__global__ void k_zero(float* __restrict__ p, int n) {
    int i = blockIdx.x * 256 + threadIdx.x;
    if (i < n) p[i] = 0.f;
}

// ---------------------------------------------------------------------------
// 7) 7x7 conv over 1024 channels (padding 3), channel-chunked partial sums.
//    WG tile: 16(i) x 64(j); each thread produces 4 adjacent j outputs.
//    Patch cached as f32 in LDS; 3 aligned DS loads feed 28 FMAs per (c,dy).
// ---------------------------------------------------------------------------
#define CONV_CC 4
__global__ __launch_bounds__(256)
void k_conv7(const unsigned short* __restrict__ hbuf,
             const float* __restrict__ w2,
             float* __restrict__ accum) {
    __shared__ float patch[CONV_CC][22][72];   // 22x70 halo, row-padded to 72
    __shared__ float wsw[CONV_CC][49];

    const int tid = threadIdx.x;
    const int tx = tid & 15;        // j-group (4 outputs each)
    const int ty = tid >> 4;        // i row 0..15
    const int jb = blockIdx.x * 64;
    const int ib = blockIdx.y * 16;
    const int c0 = blockIdx.z * 128;

    float acc[4] = {0.f, 0.f, 0.f, 0.f};

    for (int cc = 0; cc < 128; cc += CONV_CC) {
        for (int e = tid; e < CONV_CC * 49; e += 256)
            wsw[e / 49][e % 49] = w2[(size_t)(c0 + cc + e / 49) * 49 + (e % 49)];
        for (int e = tid; e < CONV_CC * 22 * 70; e += 256) {
            const int c = e / (22 * 70), r = e % (22 * 70);
            const int pi = r / 70, pj = r % 70;
            const int gi = ib + pi - 3, gj = jb + pj - 3;
            float v = 0.f;
            if ((unsigned)gi < 256u && (unsigned)gj < 256u)
                v = bf2f(hbuf[(size_t)(c0 + cc + c) * 65536 + (size_t)gi * 256 + gj]);
            patch[c][pi][pj] = v;
        }
        __syncthreads();
#pragma unroll
        for (int c = 0; c < CONV_CC; ++c) {
#pragma unroll
            for (int dy = 0; dy < 7; ++dy) {
                const float* row = &patch[c][ty + dy][tx * 4];
                float4 rv0 = *(const float4*)(row);       // 16B-aligned DS loads
                float4 rv1 = *(const float4*)(row + 4);
                float2 rv2 = *(const float2*)(row + 8);
                float r[10] = {rv0.x, rv0.y, rv0.z, rv0.w,
                               rv1.x, rv1.y, rv1.z, rv1.w,
                               rv2.x, rv2.y};
                const float* wrow = &wsw[c][dy * 7];
#pragma unroll
                for (int dx = 0; dx < 7; ++dx) {
                    const float w = wrow[dx];
                    acc[0] += r[dx + 0] * w;
                    acc[1] += r[dx + 1] * w;
                    acc[2] += r[dx + 2] * w;
                    acc[3] += r[dx + 3] * w;
                }
            }
        }
        __syncthreads();
    }
    const int oi = ib + ty;
#pragma unroll
    for (int q = 0; q < 4; ++q)
        atomicAdd(&accum[(size_t)oi * 256 + (jb + tx * 4 + q)], acc[q]);
}

// ---------------------------------------------------------------------------
// 8) sigmoid -> maps
// ---------------------------------------------------------------------------
__global__ void k_sig(const float* __restrict__ accum, const float* __restrict__ b,
                      float* __restrict__ out) {
    int i = blockIdx.x * 256 + threadIdx.x;
    float v = accum[i] + b[0];
    out[i] = 1.f / (1.f + __expf(-v));
}

// ---------------------------------------------------------------------------
// Host-side launch
// ---------------------------------------------------------------------------
extern "C" void kernel_launch(void* const* d_in, const int* in_sizes, int n_in,
                              void* d_out, int out_size, void* d_ws, size_t ws_size,
                              hipStream_t stream) {
    (void)in_sizes; (void)n_in; (void)out_size; (void)ws_size;

    const float* attentions = (const float*)d_in[0];
    const float* pooled     = (const float*)d_in[1];
    const int*   attn_mask  = (const int*)  d_in[2];
    const float* lin1_w     = (const float*)d_in[3];
    const float* lin1_b     = (const float*)d_in[4];
    const float* cls_w      = (const float*)d_in[5];
    const float* cls_b      = (const float*)d_in[6];
    const float* conv1_w    = (const float*)d_in[7];
    const float* conv1_b    = (const float*)d_in[8];
    const float* conv2_w    = (const float*)d_in[9];
    const float* conv2_b    = (const float*)d_in[10];

    char* ws = (char*)d_ws;
    float*          t_buf    = (float*)(ws + 0);                 //   4 KB
    int*            mask_eff = (int*)  (ws + 4096);              //   1 KB
    float*          accum    = (float*)(ws + 8192);              // 256 KB
    unsigned short* zh       = (unsigned short*)(ws + 524288);   // 512 KB
    unsigned short* cwbf     = (unsigned short*)(ws + 1048576);  //   4 MB
    unsigned short* hbuf     = (unsigned short*)(ws + 5242880);  // 128 MB

    float* logits = (float*)d_out;           // [2000]
    float* maps   = (float*)d_out + NEC_DIM; // [65536]

    // logits path + prep
    k_mask  <<<1,   256, 0, stream>>>(attn_mask, mask_eff);
    k_prep_z<<<L_SEQ, 256, 0, stream>>>(attentions, mask_eff, zh);
    k_prep_w<<<2048, 256, 0, stream>>>(conv1_w, cwbf, (C_DIM * 2 * C_DIM) / 4);
    k_gemv  <<<128, 256, 0, stream>>>(lin1_w, pooled, lin1_b, t_buf, C_DIM, C_DIM);
    k_gemv  <<<250, 256, 0, stream>>>(cls_w, t_buf, cls_b, logits, NEC_DIM, C_DIM);

    // pairwise-feature GEMM (WMMA bf16): h[1024][256][256] bf16
    dim3 gGemm(32, 32, 4);   // j-tiles, i-tiles, o-blocks(256)
    k_pairgemm<<<gGemm, 256, 0, stream>>>(zh, cwbf, conv1_b, hbuf);

    // 7x7 conv + sigmoid
    k_zero<<<256, 256, 0, stream>>>(accum, 65536);
    dim3 gConv(4, 16, 8);    // j-tiles(64), i-tiles(16), c-chunks(128)
    k_conv7<<<gConv, 256, 0, stream>>>(hbuf, conv2_w, accum);
    k_sig  <<<256, 256, 0, stream>>>(accum, conv2_b, maps);
}
